// MLA_32736240730631
// MI455X (gfx1250) — compile-verified, matrix operand order checked by reading, not run on hardware
//
#include <hip/hip_runtime.h>
#include <hip/hip_bf16.h>

// ---------------------------------------------------------------------------
// MLA forward for MI455X (gfx1250): bf16 WMMA GEMMs with TDM + async-LDS
// double-buffered staging. B=4, S=2048, E=2048, C=512, NHEAD=16, DHEAD=128.
// ---------------------------------------------------------------------------

typedef __bf16 bf16;
typedef __attribute__((ext_vector_type(16))) __bf16 v16bf;
typedef __attribute__((ext_vector_type(8)))  __bf16 v8bf;
typedef __attribute__((ext_vector_type(8)))  float  v8f;
typedef __attribute__((ext_vector_type(4)))  unsigned int v4u;
typedef __attribute__((ext_vector_type(8)))  int v8i_;
typedef __attribute__((ext_vector_type(4)))  int v4i_;

union FragBF { v16bf v; v8bf h[2]; };

#define BDIM 256

// ---------------------------------------------------------------------------
// f32 -> bf16 conversion
// ---------------------------------------------------------------------------
__global__ __launch_bounds__(BDIM)
void cvt_f32_bf16_kernel(const float* __restrict__ in, bf16* __restrict__ out, long n)
{
    long i = (long)blockIdx.x * BDIM + threadIdx.x;
    if (i < n) out[i] = (bf16)in[i];
}

// ---------------------------------------------------------------------------
// f32 (R x Cc) -> bf16 transpose (Cc x R)
// ---------------------------------------------------------------------------
__global__ __launch_bounds__(BDIM)
void transpose_cvt_kernel(const float* __restrict__ in, bf16* __restrict__ out,
                          int R, int Cc)
{
    long i = (long)blockIdx.x * BDIM + threadIdx.x;
    long n = (long)R * Cc;
    if (i >= n) return;
    int r = (int)(i / Cc);
    int c = (int)(i % Cc);
    out[(long)c * R + r] = (bf16)in[i];
}

// ---------------------------------------------------------------------------
// Batched bf16 GEMM: C[z] = A[z] (MxK) * op(B[z]) (+ C[z] if ACCUM)
//   BT==1 : B stored row-major (N,K)  ->  C = A * B^T
//   BT==0 : B stored row-major (K,N)  ->  C = A * B
// f32 output; optional bf16 mirror in Cb.
//
// 256 threads = 8 waves; block tile 128x128; wave tile 64x32 (4x2 WMMAs).
// Double-buffered LDS. A tile staged by the Tensor Data Mover (wave 0 issues
// one tensor_load_to_lds per tile, D# pads each 64B row by 16B to produce the
// 80B LDS row stride the fragment ds_load_b128s expect). B tile staged with
// global_load_async_to_lds_b128 when BT (coalesced), sync transpose otherwise.
// M,N multiples of 128; K multiple of 32 (true for all call sites).
// ---------------------------------------------------------------------------
#define KT 32
#define LDSP 40   // padded LDS row stride (elements); 80B keeps 16B alignment

__global__ __launch_bounds__(BDIM)
void gemm_bf16_kernel(const bf16* __restrict__ A, const bf16* __restrict__ B,
                      float* __restrict__ C, bf16* __restrict__ Cb,
                      int M, int N, int K, int BT, int ACCUM,
                      long sA, long sB, long sC)
{
    __shared__ __align__(16) __bf16 lsA[2][128][LDSP];   // 2 x 10240 B
    __shared__ __align__(16) __bf16 lsB[2][128][LDSP];   // 2 x 10240 B

    const int z = blockIdx.z;
    A += (long)z * sA;
    B += (long)z * sB;
    C += (long)z * sC;
    if (Cb) Cb += (long)z * sC;

    const int tid  = threadIdx.x;
    const int lane = tid & 31;
    const int w    = tid >> 5;     // wave 0..7
    const int wm   = w >> 2;       // 0..1 : wave row
    const int wn   = w & 3;        // 0..3 : wave col
    const long bm  = (long)blockIdx.y * 128;
    const long bn  = (long)blockIdx.x * 128;

    // zero accumulators
    v8f acc[4][2];
    {
        v8f zv;
#pragma unroll
        for (int e = 0; e < 8; ++e) zv[e] = 0.0f;
#pragma unroll
        for (int i = 0; i < 4; ++i)
#pragma unroll
            for (int j = 0; j < 2; ++j) acc[i][j] = zv;
    }

    // B tile staging assignments
    const int arow = tid >> 1;            // 0..127  (BT layout: row of B)
    const int acol = (tid & 1) * 16;      // 0 or 16
    const int bkk  = tid >> 3;            // 0..31   (NN layout)
    const int bn0  = (tid & 7) * 16;      // 0..112  (NN layout)

    const int mrow = lane & 15;
    const int kb   = (lane < 16) ? 0 : 8; // documented 16-bit A/B K split

    // ---- TDM stage of the 128xKT A tile into lsA[buf] (wave 0 only) --------
    auto stageA_tdm = [&](int kt, int buf) {
        const long k0 = (long)kt * KT;
        const unsigned long long ga =
            (unsigned long long)(const void*)(A + bm * (long)K + k0);
        const unsigned la =
            (unsigned)(unsigned long long)(const void*)&lsA[buf][0][0];
        // D# group 0: count=1, lds_addr, 57-bit global addr, type=2
        v4u g0;
        g0[0] = 1u;
        g0[1] = la;
        g0[2] = (unsigned)ga;
        g0[3] = (unsigned)((ga >> 32) & 0x1FFFFFFu) | (2u << 30);
        // D# group 1: data_size=2B, pad_enable, pad_interval=16 DW (64B rows),
        // pad_amount=4 DW (16B) -> 80B LDS row stride; tile 32x128;
        // tensor_dim0_stride = K elements; large tensor dims (never OOB).
        v8i_ g1;
        g1[0] = (int)((1u << 16) | (1u << 20) | (3u << 22) | (3u << 25));
        g1[1] = (int)(((unsigned)K & 0xFFFFu) << 16);          // tensor_dim0 lo
        g1[2] = (int)((unsigned)K >> 16);                      // dim0 hi | dim1 lo(=0)
        g1[3] = (int)((1u << 8) | (32u << 16));                // dim1 hi(=1<<24) | tile_dim0=32
        g1[4] = (int)128u;                                     // tile_dim1=128, tile_dim2=0
        g1[5] = (int)(unsigned)K;                              // dim0_stride lo
        g1[6] = 0;                                             // dim0_stride hi | dim1_stride lo
        g1[7] = 0;
        v4i_ gz;
        gz[0] = 0; gz[1] = 0; gz[2] = 0; gz[3] = 0;
        v8i_ gz8;
#pragma unroll
        for (int i = 0; i < 8; ++i) gz8[i] = 0;
        __builtin_amdgcn_tensor_load_to_lds(g0, g1, gz, gz, gz8, 0);
    };

    // ---- B tile staging ----------------------------------------------------
    auto stageB = [&](int kt, int buf) {
        const int k0 = kt * KT;
        if (BT) {
            // async global -> LDS, 16B per lane per op, 2 ops per thread
            const bf16* g0p = B + (bn + arow) * (long)K + k0 + acol;
            unsigned l0 = (unsigned)(unsigned long long)(const void*)&lsB[buf][arow][acol];
            asm volatile("global_load_async_to_lds_b128 %0, %1, off"
                         :: "v"(l0), "v"(g0p) : "memory");
            const bf16* g1p = g0p + 8;
            unsigned l1 = l0 + 16;
            asm volatile("global_load_async_to_lds_b128 %0, %1, off"
                         :: "v"(l1), "v"(g1p) : "memory");
        } else {
            // transpose-on-store: lsB[n][k] = B[k][n]
            const v8bf* gp = (const v8bf*)(B + (long)(k0 + bkk) * N + bn + bn0);
            v8bf v0 = gp[0];
            v8bf v1 = gp[1];
#pragma unroll
            for (int i = 0; i < 8; ++i) lsB[buf][bn0 + i][bkk] = v0[i];
#pragma unroll
            for (int i = 0; i < 8; ++i) lsB[buf][bn0 + 8 + i][bkk] = v1[i];
        }
    };

    const int nk = K / KT;

    // prologue: stage tile 0 into buffer 0
    if (tid < 32) stageA_tdm(0, 0);
    stageB(0, 0);

    for (int kt = 0; kt < nk; ++kt) {
        const int cur = kt & 1;
        const int nxt = cur ^ 1;

        // issue next tile while current is still in flight / being consumed
        if (kt + 1 < nk) {
            if (tid < 32) stageA_tdm(kt + 1, nxt);
            stageB(kt + 1, nxt);
        }

        // wait for tile kt only (1 TDM + 2 async ops per tile remain beyond it)
        if (tid < 32) {
            if (kt + 1 < nk) __builtin_amdgcn_s_wait_tensorcnt(1);
            else             __builtin_amdgcn_s_wait_tensorcnt(0);
        }
        if (kt + 1 < nk) asm volatile("s_wait_asynccnt 0x2" ::: "memory");
        else             asm volatile("s_wait_asynccnt 0x0" ::: "memory");
        __syncthreads();   // publish tile kt to all waves

        // ---- per-wave fragments (ds_load_b128 pairs, layout-matched)
        FragBF afr[4], bfr[2];
#pragma unroll
        for (int mt = 0; mt < 4; ++mt) {
            const int r = wm * 64 + mt * 16 + mrow;
            afr[mt].h[0] = *(const v8bf*)&lsA[cur][r][kb];
            afr[mt].h[1] = *(const v8bf*)&lsA[cur][r][kb + 16];
        }
#pragma unroll
        for (int nt = 0; nt < 2; ++nt) {
            const int c = wn * 32 + nt * 16 + mrow;
            bfr[nt].h[0] = *(const v8bf*)&lsB[cur][c][kb];
            bfr[nt].h[1] = *(const v8bf*)&lsB[cur][c][kb + 16];
        }

        // ---- 8 WMMAs per wave per K-step
#pragma unroll
        for (int mt = 0; mt < 4; ++mt)
#pragma unroll
            for (int nt = 0; nt < 2; ++nt)
                acc[mt][nt] = __builtin_amdgcn_wmma_f32_16x16x32_bf16(
                    false, afr[mt].v, false, bfr[nt].v,
                    (short)0, acc[mt][nt], false, false);

        __syncthreads();   // all reads of buffer `cur` done before it is refilled
    }

    // ---- epilogue: C/D layout -> lanes 0-15: M=v, lanes 16-31: M=8+v, N=lane&15
    const int nlo = lane & 15;
    const int mhi = (lane < 16) ? 0 : 8;
#pragma unroll
    for (int mt = 0; mt < 4; ++mt) {
#pragma unroll
        for (int nt = 0; nt < 2; ++nt) {
            const long col = bn + wn * 32 + nt * 16 + nlo;
#pragma unroll
            for (int v = 0; v < 8; ++v) {
                const long row = bm + wm * 64 + mt * 16 + mhi + v;
                const long idx = row * (long)N + col;
                float val = acc[mt][nt][v];
                if (ACCUM) val += C[idx];
                C[idx] = val;
                if (Cb) Cb[idx] = (bf16)val;
            }
        }
    }
}

// ---------------------------------------------------------------------------
// RMSNorm over rows: out = x * rsqrt(mean(x^2)+eps) * g. f32 and/or bf16 out.
// ---------------------------------------------------------------------------
__global__ __launch_bounds__(BDIM)
void rmsnorm_kernel(const float* __restrict__ x, const float* __restrict__ g,
                    float* __restrict__ outf, bf16* __restrict__ outb, int cols)
{
    __shared__ float red[8];
    const long row = blockIdx.x;
    const float* xr = x + row * (long)cols;
    const int tid = threadIdx.x;

    float ss = 0.0f;
    for (int c = tid; c < cols; c += BDIM) { float v = xr[c]; ss += v * v; }
#pragma unroll
    for (int off = 16; off; off >>= 1) ss += __shfl_xor(ss, off, 32);
    if ((tid & 31) == 0) red[tid >> 5] = ss;
    __syncthreads();
    if (tid == 0) {
        float t = 0.0f;
#pragma unroll
        for (int i = 0; i < 8; ++i) t += red[i];
        red[0] = t;
    }
    __syncthreads();
    const float scale = rsqrtf(red[0] / (float)cols + 1.1920929e-7f);

    for (int c = tid; c < cols; c += BDIM) {
        float v = xr[c] * scale * g[c];
        if (outf) outf[row * (long)cols + c] = v;
        if (outb) outb[row * (long)cols + c] = (bf16)v;
    }
}

// ---------------------------------------------------------------------------
// RoPE on flat (B*S*NHEAD, 128) rows; one thread per (even,odd) pair.
// ---------------------------------------------------------------------------
__global__ __launch_bounds__(BDIM)
void rope_kernel(const float* __restrict__ x, float* __restrict__ outf,
                 bf16* __restrict__ outb, long npairs, int S)
{
    const long p = (long)blockIdx.x * BDIM + threadIdx.x;
    if (p >= npairs) return;
    const int  d    = (int)(p & 63);          // pair index within head
    const long rowh = p >> 6;                 // (b*S+s)*NHEAD + h
    const int  s    = (int)((rowh >> 4) & (long)(S - 1));

    // freq = 10000^(-2d/128) = exp(-(2d/128)*ln(10000))
    const float freq = __expf(-((2.0f * d) / 128.0f) * 9.210340371976184f);
    const float ang  = (float)s * freq;
    float sn, cs;
    __sincosf(ang, &sn, &cs);

    const long base = rowh * 128 + 2 * d;
    const float x1 = x[base], x2 = x[base + 1];
    const float o1 = x1 * cs - x2 * sn;
    const float o2 = x1 * sn + x2 * cs;
    if (outf) { outf[base] = o1; outf[base + 1] = o2; }
    if (outb) { outb[base] = (bf16)o1; outb[base + 1] = (bf16)o2; }
}

// ---------------------------------------------------------------------------
// Row softmax with scale; bf16 output (feeds attn @ ckv WMMA GEMM).
// ---------------------------------------------------------------------------
__global__ __launch_bounds__(BDIM)
void softmax_kernel(const float* __restrict__ s, bf16* __restrict__ out,
                    int cols, float scale)
{
    __shared__ float red[8];
    const long row = blockIdx.x;
    const float* xr = s + row * (long)cols;
    const int tid = threadIdx.x;

    float mx = -3.4e38f;
    for (int c = tid; c < cols; c += BDIM) mx = fmaxf(mx, xr[c] * scale);
#pragma unroll
    for (int off = 16; off; off >>= 1) mx = fmaxf(mx, __shfl_xor(mx, off, 32));
    if ((tid & 31) == 0) red[tid >> 5] = mx;
    __syncthreads();
    if (tid == 0) {
        float m = red[0];
#pragma unroll
        for (int i = 1; i < 8; ++i) m = fmaxf(m, red[i]);
        red[0] = m;
    }
    __syncthreads();
    mx = red[0];
    __syncthreads();

    float sum = 0.0f;
    for (int c = tid; c < cols; c += BDIM) sum += __expf(xr[c] * scale - mx);
#pragma unroll
    for (int off = 16; off; off >>= 1) sum += __shfl_xor(sum, off, 32);
    if ((tid & 31) == 0) red[tid >> 5] = sum;
    __syncthreads();
    if (tid == 0) {
        float t = 0.0f;
#pragma unroll
        for (int i = 0; i < 8; ++i) t += red[i];
        red[0] = t;
    }
    __syncthreads();
    const float inv = 1.0f / red[0];

    for (int c = tid; c < cols; c += BDIM)
        out[row * (long)cols + c] = (bf16)(__expf(xr[c] * scale - mx) * inv);
}

// ---------------------------------------------------------------------------
// Host orchestration
// ---------------------------------------------------------------------------
static inline long cdiv(long a, long b) { return (a + b - 1) / b; }

extern "C" void kernel_launch(void* const* d_in, const int* in_sizes, int n_in,
                              void* d_out, int out_size, void* d_ws, size_t ws_size,
                              hipStream_t stream)
{
    (void)in_sizes; (void)n_in; (void)out_size; (void)ws_size;

    constexpr int  B = 4, S = 2048, E = 2048, C = 512;
    constexpr long MROWS = (long)B * S;                 // 8192
    const float softmax_scale = 1.0f / 12.0f;           // 1/sqrt(16+128)

    const float* h      = (const float*)d_in[0];
    const float* w_d_kv = (const float*)d_in[1];
    const float* w_u_k  = (const float*)d_in[2];
    const float* w_u_v  = (const float*)d_in[3];
    const float* w_d_q  = (const float*)d_in[4];
    const float* w_u_q  = (const float*)d_in[5];
    const float* w_kr   = (const float*)d_in[6];
    const float* w_qr   = (const float*)d_in[7];
    const float* w_o    = (const float*)d_in[8];
    const float* g1     = (const float*)d_in[9];
    const float* g2     = (const float*)d_in[10];
    const float* g3     = (const float*)d_in[11];

    float* out_u   = (float*)d_out;                     // (B,S,E)
    float* out_ckv = out_u + MROWS * E;                 // (B,S,C)
    float* out_kr  = out_ckv + MROWS * C;               // (B,S,E)

    // ---- workspace carve
    size_t off = 0;
    auto carve = [&](size_t bytes) -> void* {
        void* p = (char*)d_ws + off;
        off += (bytes + 255) & ~(size_t)255;
        return p;
    };
    bf16* hb       = (bf16*)carve(MROWS * E * 2);       // h bf16
    bf16* wdkv_b   = (bf16*)carve((long)C * E * 2);
    bf16* wuk_b    = (bf16*)carve((long)E * C * 2);
    bf16* wuv_b    = (bf16*)carve((long)E * C * 2);
    bf16* wdq_b    = (bf16*)carve((long)C * E * 2);
    bf16* wuq_b    = (bf16*)carve((long)E * C * 2);
    bf16* wkr_b    = (bf16*)carve((long)E * E * 2);
    bf16* wqr_b    = (bf16*)carve((long)E * C * 2);
    bf16* wo_b     = (bf16*)carve((long)E * E * 2);
    bf16* cq_b     = (bf16*)carve(MROWS * C * 2);
    bf16* ckv_b    = (bf16*)carve(MROWS * C * 2);
    bf16* qr_b     = (bf16*)carve(MROWS * E * 2);
    bf16* kr_b     = (bf16*)carve(MROWS * E * 2);
    bf16* aq_b     = (bf16*)carve(MROWS * C * 2);
    bf16* attn_b   = (bf16*)carve(MROWS * (long)S * 2);
    bf16* o_b      = (bf16*)carve(MROWS * C * 2);
    bf16* t2t_b    = (bf16*)carve((long)E * E * 2);
    bf16* absq_b   = (bf16*)carve((long)E * C * 2);
    bf16* awo_b    = (bf16*)carve((long)C * E * 2);
    float* R_pre   = (float*)carve(MROWS * E * 4);      // serial f32 arena #1
    float* R_big   = (float*)carve(MROWS * (long)S * 4);// serial f32 arena #2

    auto cvt = [&](const float* src, bf16* dst, long n) {
        cvt_f32_bf16_kernel<<<dim3((unsigned)cdiv(n, BDIM)), BDIM, 0, stream>>>(src, dst, n);
    };
    auto gemm = [&](const bf16* A, const bf16* Bm, float* Cm, bf16* Cb,
                    int M, int N, int K, int BT, int ACCUM,
                    long sA, long sB, long sC, int Z) {
        dim3 grid((unsigned)(N / 128), (unsigned)(M / 128), (unsigned)Z);
        gemm_bf16_kernel<<<grid, BDIM, 0, stream>>>(A, Bm, Cm, Cb, M, N, K, BT, ACCUM, sA, sB, sC);
    };

    // ---- 0) precision conversion (f32 -> bf16)
    cvt(h,      hb,     MROWS * E);
    cvt(w_d_kv, wdkv_b, (long)C * E);
    cvt(w_u_k,  wuk_b,  (long)E * C);
    cvt(w_u_v,  wuv_b,  (long)E * C);
    cvt(w_d_q,  wdq_b,  (long)C * E);
    cvt(w_u_q,  wuq_b,  (long)E * C);
    cvt(w_kr,   wkr_b,  (long)E * E);
    cvt(w_qr,   wqr_b,  (long)E * C);
    cvt(w_o,    wo_b,   (long)E * E);

    // ---- 1) cq = rmsnorm(h @ w_d_q^T, g1)           (bf16 only)
    gemm(hb, wdq_b, R_pre, nullptr, (int)MROWS, C, E, /*BT=*/1, 0, 0, 0, 0, 1);
    rmsnorm_kernel<<<dim3((unsigned)MROWS), BDIM, 0, stream>>>(R_pre, g1, nullptr, cq_b, C);

    // ---- 2) ckv = rmsnorm(h @ w_d_kv^T, g2)         (f32 output + bf16)
    gemm(hb, wdkv_b, R_pre, nullptr, (int)MROWS, C, E, 1, 0, 0, 0, 0, 1);
    rmsnorm_kernel<<<dim3((unsigned)MROWS), BDIM, 0, stream>>>(R_pre, g2, out_ckv, ckv_b, C);

    // ---- 3) qr = rope(cq @ w_qr^T)                  (bf16 only)
    gemm(cq_b, wqr_b, R_pre, nullptr, (int)MROWS, E, C, 1, 0, 0, 0, 0, 1);
    rope_kernel<<<dim3((unsigned)cdiv(MROWS * E / 2, BDIM)), BDIM, 0, stream>>>(
        R_pre, nullptr, qr_b, MROWS * E / 2, S);

    // ---- 4) kr = rope(h @ w_kr^T)                   (f32 output + bf16)
    gemm(hb, wkr_b, R_pre, nullptr, (int)MROWS, E, E, 1, 0, 0, 0, 0, 1);
    rope_kernel<<<dim3((unsigned)cdiv(MROWS * E / 2, BDIM)), BDIM, 0, stream>>>(
        R_pre, out_kr, kr_b, MROWS * E / 2, S);

    // ---- 5) absorbed_w_q = (w_u_q @ w_d_q)^T @ w_u_k   ((E,C), bf16)
    gemm(wuq_b, wdq_b, R_pre, nullptr, E, E, C, /*BT=*/0, 0, 0, 0, 0, 1);     // t2 = (E,E)
    transpose_cvt_kernel<<<dim3((unsigned)cdiv((long)E * E, BDIM)), BDIM, 0, stream>>>(
        R_pre, t2t_b, E, E);                                                  // t2^T bf16
    gemm(t2t_b, wuk_b, R_big, absq_b, E, C, E, 0, 0, 0, 0, 0, 1);             // absorbed_q

    // ---- 6) aq = h @ absorbed_w_q                    ((B*S,C), bf16)
    gemm(hb, absq_b, R_pre, aq_b, (int)MROWS, C, E, 0, 0, 0, 0, 0, 1);

    // ---- 7) absorbed_w_o = (w_o @ w_u_v)^T           ((C,E), bf16)
    gemm(wo_b, wuv_b, R_pre, nullptr, E, C, E, 0, 0, 0, 0, 0, 1);             // t3 = (E,C)
    transpose_cvt_kernel<<<dim3((unsigned)cdiv((long)E * C, BDIM)), BDIM, 0, stream>>>(
        R_pre, awo_b, E, C);

    // ---- 8) scores = aq @ ckv^T + qr @ kr^T          (batched over B)
    gemm(aq_b, ckv_b, R_big, nullptr, S, S, C, /*BT=*/1, /*ACCUM=*/0,
         (long)S * C, (long)S * C, (long)S * S, B);
    gemm(qr_b, kr_b, R_big, nullptr, S, S, E, 1, /*ACCUM=*/1,
         (long)S * E, (long)S * E, (long)S * S, B);

    // ---- 9) attn = softmax(scores / sqrt(144))       (bf16)
    softmax_kernel<<<dim3((unsigned)MROWS), BDIM, 0, stream>>>(R_big, attn_b, S, softmax_scale);

    // ---- 10) o = attn @ ckv                          (batched, bf16)
    gemm(attn_b, ckv_b, R_pre, o_b, S, C, S, /*BT=*/0, 0,
         (long)S * S, (long)S * C, (long)S * C, B);

    // ---- 11) u = rmsnorm(o @ absorbed_w_o, g3)       (f32 output)
    gemm(o_b, awo_b, R_big, nullptr, (int)MROWS, E, C, 0, 0, 0, 0, 0, 1);
    rmsnorm_kernel<<<dim3((unsigned)MROWS), BDIM, 0, stream>>>(R_big, g3, out_u, nullptr, E);
}
